// VGG16_add_partial_60138132079270
// MI455X (gfx1250) — matrix operand order repaired and checked by reading, not compiled, not run on hardware
//
#include <hip/hip_runtime.h>

typedef _Float16 v16h_t __attribute__((ext_vector_type(16)));
typedef _Float16 v8h_t  __attribute__((ext_vector_type(8)));
typedef float    v8f_t  __attribute__((ext_vector_type(8)));

// -------------------------------------------------------------------------
// Pack fp32 NCHW input [128,3,32,32] -> f16 NHWC [128,32,32,32] (zero-padded
// channels 3..31 so every conv's K dimension is a multiple of 32).
// -------------------------------------------------------------------------
__global__ void vgg_pack_input(const float* __restrict__ x, _Float16* __restrict__ out) {
    int id = blockIdx.x * blockDim.x + threadIdx.x;  // exact grid: 128*32*32*32
    int c   = id & 31;
    int pix = id >> 5;
    int xx  = pix & 31;
    int rs  = pix >> 5;
    int yy  = rs & 31;
    int b   = rs >> 5;
    float v = (c < 3) ? x[(((b * 3 + c) * 32) + yy) * 32 + xx] : 0.0f;
    out[id] = (_Float16)v;
}

// -------------------------------------------------------------------------
// Pack fp32 OIHW weights into f16 WMMA B-fragment order.
// Blocks of 512 halves: block = ((t*kChunks + kc)*nTilesN + nc),
// within block: idx = lane*16 + e, where K_local = (lane>>4)*16 + e (0..31),
// N_local = lane & 15. Zero-pads K >= Cin and N >= CoutActual.
// All divisors are powers of two -> shifts/masks (logNtN, logKC).
// -------------------------------------------------------------------------
__global__ void vgg_pack_weights(const float* __restrict__ w, _Float16* __restrict__ dst,
                                 int Cin, int CoutAct, int ksize,
                                 int logNtN, int logKC) {
    int id = blockIdx.x * blockDim.x + threadIdx.x;  // exact grid: taps*Cpad*Cout
    int e     = id & 15;
    int lane  = (id >> 4) & 31;
    int block = id >> 9;
    int nc  = block & ((1 << logNtN) - 1);
    int tkc = block >> logNtN;
    int kc  = tkc & ((1 << logKC) - 1);
    int t   = tkc >> logKC;
    int ky  = t / ksize;
    int kx  = t - ky * ksize;
    int K = (kc << 5) + ((lane >> 4) << 4) + e;  // input channel
    int n = (nc << 4) + (lane & 15);             // output channel
    float v = 0.0f;
    if (K < Cin && n < CoutAct)
        v = w[(((long)n * Cin + K) * ksize + ky) * ksize + kx];
    dst[id] = (_Float16)v;
}

__device__ __forceinline__ float vgg_act(float v, int mode) {
    if (mode == 0)      return fminf(fmaxf(v, 0.0f), 1.0f);
    else if (mode == 1) return fmaxf(v, 0.0f);
    return v;
}

// -------------------------------------------------------------------------
// Main 3x3 conv kernel: implicit GEMM via V_WMMA_F32_16X16X32_F16.
// Block = 4 waves = 4 M-tiles (64 pixels) x one 64-channel N-group.
// Per K-stage s = t*kChunks+kc: the 4 KB B macro-tile (shared by all 4
// waves) is staged into LDS with double-buffered
// GLOBAL_LOAD_ASYNC_TO_LDS_B128 (ASYNCcnt), then each wave does one
// A-fragment load + 4 DS fragment reads + 4 WMMAs.
// All geometry is power-of-two: logW (H==W), logKC (kChunks=Cpad/32),
// logCout, logNtG (ntGroups=Cout/64). mode: 0=clip01, 1=relu.
// -------------------------------------------------------------------------
__global__ __launch_bounds__(128)
void vgg_conv_wmma_tile(const _Float16* __restrict__ in, const _Float16* __restrict__ wfrag,
                        const float* __restrict__ scale, const float* __restrict__ bias,
                        _Float16* __restrict__ out,
                        int logW, int logKC, int logCout, int logNtG, int mode) {
    __shared__ __align__(16) _Float16 smem[2][4 * 512];   // 2 x 4KB double buffer

    const int tid     = threadIdx.x;
    const int wave    = tid >> 5;
    const int lane    = tid & 31;
    const int lane16  = lane & 15;
    const int halfSel = lane >> 4;

    const int W     = 1 << logW;                 // H == W
    const int kcMask = (1 << logKC) - 1;
    const int logNtN = logCout - 4;              // nTilesN = Cout/16

    const int mBlock = blockIdx.x >> logNtG;
    const int ng     = blockIdx.x & ((1 << logNtG) - 1);   // 64-ch output group
    const int mtile  = (mBlock << 2) + wave;

    // pixel owned by this lane's A rows (all power-of-two decode)
    const int p   = (mtile << 4) + lane16;
    const int b   = p >> (2 * logW);
    const int rem = p & ((1 << (2 * logW)) - 1);
    const int y   = rem >> logW;
    const int x   = rem & (W - 1);

    const int T = 9 << logKC;                    // taps(9) * kChunks stages

    v8f_t acc[4] = {{}, {}, {}, {}};

    // ---- async stage of one 4KB B macro-tile: each thread moves 32 bytes ----
    auto stage = [&](int s, int parity) {
        const long base = ((((long)s << logNtN) + (ng << 2)) << 9) + (tid << 4);
        unsigned ldsAddr = (unsigned)(unsigned long long)&smem[parity][tid * 16];
        unsigned long long g = (unsigned long long)(const void*)(wfrag + base);
        asm volatile("global_load_async_to_lds_b128 %0, %1, off"
                     :: "v"(ldsAddr), "v"(g) : "memory");
        asm volatile("global_load_async_to_lds_b128 %0, %1, off"
                     :: "v"(ldsAddr + 16u), "v"(g + 16ull) : "memory");
    };

    stage(0, 0);

    for (int s = 0; s < T; ++s) {
        const int t      = s >> logKC;
        const int kc     = s & kcMask;
        const int parity = s & 1;

        // prefetch next stage into the other buffer (its last readers synced
        // at the end of iteration s-1), then retire stage s (in-order).
        if (s + 1 < T) {
            stage(s + 1, parity ^ 1);
            asm volatile("s_wait_asynccnt 0x2" ::: "memory");
        } else {
            asm volatile("s_wait_asynccnt 0x0" ::: "memory");
        }
        __syncthreads();

        // ---- A fragment: lanes 0-15 K=0..15, lanes 16-31 K=16..31
        const int ky = t / 3;
        const int kx = t - ky * 3;
        const int yy = y + ky - 1;
        const int xx = x + kx - 1;
        const bool valid = (yy >= 0) & (yy < W) & (xx >= 0) & (xx < W);
        v8h_t alo = {}, ahi = {};
        if (valid) {
            const int  rowIdx = (((b << logW) + yy) << logW) + xx;
            const long aoff   = ((long)rowIdx << (logKC + 5)) + (kc << 5) + (halfSel << 4);
            const _Float16* ap = in + aoff;
            alo = *(const v8h_t*)(ap);
            ahi = *(const v8h_t*)(ap + 8);
        }
        v16h_t a;
#pragma unroll
        for (int e = 0; e < 8; ++e) { a[e] = alo[e]; a[e + 8] = ahi[e]; }

        // ---- 4 B fragments from LDS, 4 WMMAs sharing the A fragment ----
#pragma unroll
        for (int nb = 0; nb < 4; ++nb) {
            const v8h_t* bp = (const v8h_t*)&smem[parity][nb * 512 + lane * 16];
            v8h_t blo = bp[0];
            v8h_t bhi = bp[1];
            v16h_t bb;
#pragma unroll
            for (int e = 0; e < 8; ++e) { bb[e] = blo[e]; bb[e + 8] = bhi[e]; }
            acc[nb] = __builtin_amdgcn_wmma_f32_16x16x32_f16(
                false, a, false, bb, (short)0, acc[nb], false, false);
        }

        __syncthreads();   // all waves done reading buf[parity] before s+2 overwrites it
    }

    // ---- epilogue: BN affine + activation, scatter per C/D layout ----
#pragma unroll
    for (int nb = 0; nb < 4; ++nb) {
        const int n = ((ng << 2) + nb) * 16 + lane16;
        const float s  = scale[n];
        const float bi = bias[n];
#pragma unroll
        for (int i = 0; i < 8; ++i) {
            const int pp = (mtile << 4) + (halfSel << 3) + i;
            out[((long)pp << logCout) + n] = (_Float16)vgg_act(acc[nb][i] * s + bi, mode);
        }
    }
}

// -------------------------------------------------------------------------
// FC layer: batch GEMM [128,512] x [512,16] -> f16 [128,16] via WMMA.
// One wave per 16-row tile; K = 512 = 16 chunks of 32. No activation.
// -------------------------------------------------------------------------
__global__ __launch_bounds__(128)
void vgg_fc_wmma(const _Float16* __restrict__ in, const _Float16* __restrict__ wfrag,
                 _Float16* __restrict__ out) {
    const int wave    = threadIdx.x >> 5;
    const int lane    = threadIdx.x & 31;
    const int lane16  = lane & 15;
    const int halfSel = lane >> 4;

    const int mtile = (blockIdx.x << 2) + wave;   // 8 tiles total
    const int p     = (mtile << 4) + lane16;      // batch row

    v8f_t acc = {};
#pragma unroll
    for (int kc = 0; kc < 16; ++kc) {
        const _Float16* ap = in + ((long)p << 9) + (kc << 5) + (halfSel << 4);
        v8h_t alo = *(const v8h_t*)(ap);
        v8h_t ahi = *(const v8h_t*)(ap + 8);
        v16h_t a;
#pragma unroll
        for (int e = 0; e < 8; ++e) { a[e] = alo[e]; a[e + 8] = ahi[e]; }

        const _Float16* wb = wfrag + (kc << 9) + (lane << 4);
        v8h_t blo = *(const v8h_t*)(wb);
        v8h_t bhi = *(const v8h_t*)(wb + 8);
        v16h_t bb;
#pragma unroll
        for (int e = 0; e < 8; ++e) { bb[e] = blo[e]; bb[e + 8] = bhi[e]; }

        acc = __builtin_amdgcn_wmma_f32_16x16x32_f16(
            false, a, false, bb, (short)0, acc, false, false);
    }

#pragma unroll
    for (int i = 0; i < 8; ++i) {
        const int pp = (mtile << 4) + (halfSel << 3) + i;
        out[(pp << 4) + lane16] = (_Float16)acc[i];
    }
}

// -------------------------------------------------------------------------
// 2x2 average pool + Hardtanh(0,1) on NHWC f16. Power-of-two geometry.
// -------------------------------------------------------------------------
__global__ void vgg_pool_clip(const _Float16* __restrict__ in, _Float16* __restrict__ out,
                              int logW, int logC) {
    int id = blockIdx.x * blockDim.x + threadIdx.x;  // exact grid: B*(W/2)^2*C
    const int W = 1 << logW;
    const int logWo = logW - 1;
    int c  = id & ((1 << logC) - 1);
    int r  = id >> logC;
    int xo = r & ((1 << logWo) - 1); r >>= logWo;
    int yo = r & ((1 << logWo) - 1);
    int b  = r >> logWo;
    long base = ((((long)((b << logW) + (yo << 1)) << logW) + (xo << 1)) << logC) + c;
    long rowS = (long)W << logC;
    long colS = 1L << logC;
    float v = (float)in[base] + (float)in[base + colS] +
              (float)in[base + rowS] + (float)in[base + rowS + colS];
    v = fminf(fmaxf(v * 0.25f, 0.0f), 1.0f);
    out[id] = (_Float16)v;
}

// -------------------------------------------------------------------------
// Final affine: f16 [128,16] -> fp32 [128,10] with bn_scales[13]/bn_biases[13].
// -------------------------------------------------------------------------
__global__ void vgg_finalize(const _Float16* __restrict__ h, const float* __restrict__ s,
                             const float* __restrict__ bi, float* __restrict__ out) {
    int id = blockIdx.x * blockDim.x + threadIdx.x;
    if (id >= 128 * 10) return;
    int j = id % 10;
    int b = id / 10;
    out[id] = (float)h[(b << 4) + j] * s[j] + bi[j];
}

// -------------------------------------------------------------------------
// Host-side orchestration
// -------------------------------------------------------------------------
struct LayerCfg {
    int H, Cin, Cpad, Cout, CoutAct, ksize, mode, poolAfter;
};

static const LayerCfg g_layers[14] = {
    {32,   3,  32,  64,  64, 3, 0, 0},
    {32,  64,  64,  64,  64, 3, 1, 1},
    {16,  64,  64, 128, 128, 3, 0, 0},
    {16, 128, 128, 128, 128, 3, 1, 1},
    { 8, 128, 128, 256, 256, 3, 0, 0},
    { 8, 256, 256, 256, 256, 3, 0, 0},
    { 8, 256, 256, 256, 256, 3, 1, 1},
    { 4, 256, 256, 512, 512, 3, 0, 0},
    { 4, 512, 512, 512, 512, 3, 0, 0},
    { 4, 512, 512, 512, 512, 3, 1, 1},
    { 2, 512, 512, 512, 512, 3, 0, 0},
    { 2, 512, 512, 512, 512, 3, 0, 0},
    { 2, 512, 512, 512, 512, 3, 1, 1},
    { 1, 512, 512,  16,  10, 1, 2, 0},
};

static inline int ilog2(int v) {
    int l = 0;
    while ((1 << l) < v) ++l;
    return l;
}

extern "C" void kernel_launch(void* const* d_in, const int* in_sizes, int n_in,
                              void* d_out, int out_size, void* d_ws, size_t ws_size,
                              hipStream_t stream) {
    (void)in_sizes; (void)n_in; (void)out_size; (void)ws_size;
    const int B = 128;

    const float* x = (const float*)d_in[0];
    char* ws = (char*)d_ws;

    // ---- workspace layout ----
    size_t off = 0;
    size_t wOff[14];
    for (int i = 0; i < 14; ++i) {
        const LayerCfg& L = g_layers[i];
        wOff[i] = off;
        size_t halves = (size_t)L.ksize * L.ksize * L.Cpad * L.Cout;
        off += halves * sizeof(_Float16);
        off = (off + 255) & ~(size_t)255;
    }
    const size_t actBytes = (size_t)B * 32 * 32 * 64 * sizeof(_Float16);  // peak
    _Float16* actA = (_Float16*)(ws + off); off += actBytes;
    off = (off + 255) & ~(size_t)255;
    _Float16* actB = (_Float16*)(ws + off);

    // ---- pack input ----
    {
        long total = (long)B * 32 * 32 * 32;
        vgg_pack_input<<<(int)(total / 256), 256, 0, stream>>>(x, actA);
    }

    // ---- pack all weights into fragment order ----
    for (int i = 0; i < 14; ++i) {
        const LayerCfg& L = g_layers[i];
        long total = (long)L.ksize * L.ksize * L.Cpad * L.Cout;  // multiple of 512
        vgg_pack_weights<<<(int)(total / 256), 256, 0, stream>>>(
            (const float*)d_in[1 + i], (_Float16*)(ws + wOff[i]),
            L.Cin, L.CoutAct, L.ksize, ilog2(L.Cout) - 4, ilog2(L.Cpad) - 5);
    }

    // ---- conv/pool pipeline ----
    _Float16* cur = actA;
    _Float16* oth = actB;
    for (int i = 0; i < 13; ++i) {
        const LayerCfg& L = g_layers[i];
        const int logW    = ilog2(L.H);
        const int logKC   = ilog2(L.Cpad) - 5;
        const int logCout = ilog2(L.Cout);
        const int logNtG  = logCout - 6;                  // Cout/64 groups
        const int mTiles  = (B * L.H * L.H) >> 4;
        const int blocks  = (mTiles / 4) << logNtG;       // exact
        vgg_conv_wmma_tile<<<blocks, 128, 0, stream>>>(
            cur, (const _Float16*)(ws + wOff[i]),
            (const float*)d_in[15 + i], (const float*)d_in[29 + i], oth,
            logW, logKC, logCout, logNtG, L.mode);
        if (L.poolAfter) {
            long total = (long)B * (L.H / 2) * (L.H / 2) * L.Cout;  // multiple of 256
            vgg_pool_clip<<<(int)(total / 256), 256, 0, stream>>>(
                oth, cur, logW, logCout);
        } else {
            _Float16* t = cur; cur = oth; oth = t;
        }
    }

    // ---- FC 512 -> 16(10 used) then final affine into fp32 d_out ----
    vgg_fc_wmma<<<2, 128, 0, stream>>>(cur, (const _Float16*)(ws + wOff[13]), oth);
    vgg_finalize<<<5, 256, 0, stream>>>(
        oth, (const float*)d_in[28], (const float*)d_in[42], (float*)d_out);
}
